// HungarianMatcher_36120674959433
// MI455X (gfx1250) — matrix operand order repaired and checked by reading, not compiled, not run on hardware
//
#include <hip/hip_runtime.h>
#include <stdint.h>
#include <stddef.h>

#define B_  8
#define Q_  500
#define T_  200
#define HW_ 65536

// ---------- WMMA types ----------
typedef __attribute__((ext_vector_type(16))) __bf16 v16bf;
typedef __attribute__((ext_vector_type(8)))  float  v8f;

struct __attribute__((aligned(16))) U128 { unsigned a, b, c, d; };
struct U256 { U128 lo, hi; };                       // 32B == v16bf
struct __attribute__((aligned(8)))  U64v { unsigned x, y; };

__device__ __forceinline__ unsigned short f2bf(float f) {
  unsigned u = __float_as_uint(f);
  u += 0x7FFFu + ((u >> 16) & 1u);                  // round-to-nearest-even
  return (unsigned short)(u >> 16);
}

// ============================================================
// GEMM partial: part[ks][b][t][q] = sum_{k in chunk ks}
//                   sigmoid(pred[b][q][k]) * tgt[b][t][k]
// block = 256 threads (8 waves); macro tile M=128 x N=32, K-step 64.
// Each wave: one 16-row M sub-tile x full 32 N columns (2 wmma frags).
// K is split across blockIdx.z to fill the machine (deterministic,
// fixed-order reduction afterwards -- no float atomics).
// ============================================================
#define MT  128
#define NT  32
#define KT  64
#define LDP 72   // padded LDS row (bf16 elems): 144B stride -> conflict-free b128 frags

__global__ __launch_bounds__(256)
void mask_cost_wmma(const float* __restrict__ pred,
                    const float* __restrict__ tgt,
                    float* __restrict__ parts,
                    int ksplit, int kchunk) {
  __shared__ unsigned short As[MT * LDP];
  __shared__ unsigned short Bs[NT * LDP];

  const int tid  = threadIdx.x;
  const int lane = tid & 31;
  const int wave = tid >> 5;            // 0..7 -> M sub-tile
  const int qb   = blockIdx.x * MT;
  const int tb   = blockIdx.y * NT;
  const int bz   = blockIdx.z;
  const int b    = bz / ksplit;
  const int ks   = bz % ksplit;

  const int kbeg = ks * kchunk;
  int kend = kbeg + kchunk; if (kend > HW_) kend = HW_;

  const int ml = lane & 15;             // row-in-tile / N column
  const int kh = lane >> 4;             // lane half selects K group

  v8f c0 = {}; v8f c1 = {};

  const size_t predBase = ((size_t)b * Q_) * HW_;
  const size_t tgtBase  = ((size_t)b * T_) * HW_;

  for (int k0 = kbeg; k0 < kend; k0 += KT) {
    // ---- stage A tile: sigmoid(pred) -> bf16, MT x KT ----
    #pragma unroll
    for (int it = 0; it < (MT * KT / 4) / 256; ++it) {
      int idx4 = it * 256 + tid;
      int r  = idx4 >> 4;               // 16 float4 per 64-wide row
      int c4 = idx4 & 15;
      int q  = qb + r; if (q > Q_ - 1) q = Q_ - 1;   // clamp (dup rows, discarded at store)
      const float* gp = pred + predBase + (size_t)q * HW_ + k0 + c4 * 4;
      const float4 f = *(const float4*)gp;
      __builtin_prefetch(gp + 2 * KT, 0, 3);        // global_prefetch_b8, 2 K-tiles ahead
      float s0 = 1.0f / (1.0f + __expf(-f.x));
      float s1 = 1.0f / (1.0f + __expf(-f.y));
      float s2 = 1.0f / (1.0f + __expf(-f.z));
      float s3 = 1.0f / (1.0f + __expf(-f.w));
      U64v p2;
      p2.x = (unsigned)f2bf(s0) | ((unsigned)f2bf(s1) << 16);
      p2.y = (unsigned)f2bf(s2) | ((unsigned)f2bf(s3) << 16);
      *(U64v*)&As[r * LDP + c4 * 4] = p2;
    }
    // ---- stage B tile: tgt -> bf16, NT x KT ----
    #pragma unroll
    for (int it = 0; it < (NT * KT / 4) / 256; ++it) {
      int idx4 = it * 256 + tid;
      int r  = idx4 >> 4;
      int c4 = idx4 & 15;
      int t  = tb + r; if (t > T_ - 1) t = T_ - 1;
      const float* gp = tgt + tgtBase + (size_t)t * HW_ + k0 + c4 * 4;
      const float4 f = *(const float4*)gp;
      __builtin_prefetch(gp + 2 * KT, 0, 3);
      U64v p2;
      p2.x = (unsigned)f2bf(f.x) | ((unsigned)f2bf(f.y) << 16);
      p2.y = (unsigned)f2bf(f.z) | ((unsigned)f2bf(f.w) << 16);
      *(U64v*)&Bs[r * LDP + c4 * 4] = p2;
    }
    __syncthreads();

    #pragma unroll
    for (int kk = 0; kk < KT; kk += 32) {
      // A 16x32 bf16 frag: lanes 0-15 -> K {0..15}; lanes 16-31 -> K {16..31}
      U256 fa;
      const unsigned short* ap = &As[(wave * 16 + ml) * LDP + kk + 16 * kh];
      fa.lo = *(const U128*)(ap);
      fa.hi = *(const U128*)(ap + 8);
      v16bf a = __builtin_bit_cast(v16bf, fa);

      // B 32x16 bf16 frags: lane = N col, lane-half selects K base (0/16)
      U256 fb0, fb1;
      const unsigned short* bp0 = &Bs[(0  + ml) * LDP + kk + 16 * kh];
      const unsigned short* bp1 = &Bs[(16 + ml) * LDP + kk + 16 * kh];
      fb0.lo = *(const U128*)(bp0); fb0.hi = *(const U128*)(bp0 + 8);
      fb1.lo = *(const U128*)(bp1); fb1.hi = *(const U128*)(bp1 + 8);
      v16bf b0 = __builtin_bit_cast(v16bf, fb0);
      v16bf b1 = __builtin_bit_cast(v16bf, fb1);

      c0 = __builtin_amdgcn_wmma_f32_16x16x32_bf16(false, a, false, b0, (short)0, c0, false, false);
      c1 = __builtin_amdgcn_wmma_f32_16x16x32_bf16(false, a, false, b1, (short)0, c1, false, false);
    }
    __syncthreads();
  }

  // epilogue: D layout -> lane l, vgpr i: M = i + 8*(l>>4), N = l&15
  // Always store (zeros if empty K-range) so every partial slot is defined.
  float* outP = parts + (size_t)ks * ((size_t)B_ * T_ * Q_);
  const int q0 = qb + wave * 16;
  #pragma unroll
  for (int i = 0; i < 8; ++i) {
    int q = q0 + i + 8 * kh;
    if (q < Q_) {
      int t0 = tb + ml;
      if (t0 < T_)      outP[((size_t)b * T_ + t0)      * Q_ + q] = c0[i];
      if (t0 + 16 < T_) outP[((size_t)b * T_ + t0 + 16) * Q_ + q] = c1[i];
    }
  }
}

// ============================================================
// Fixed-order reduction over K-split partials, with negation:
// costT[i] = -(sum_s parts[s][i]).  In-place safe when costT==parts[0]
// (ksplit==1): each element is read before written by the same thread.
// ============================================================
__global__ __launch_bounds__(256)
void reduce_negate(const float* __restrict__ parts,
                   float* __restrict__ costT, int ksplit) {
  const size_t S  = (size_t)B_ * T_ * Q_;       // elements per partial
  const size_t N4 = S / 4;                      // 800000/4 = 200000
  for (size_t i = (size_t)blockIdx.x * blockDim.x + threadIdx.x;
       i < N4; i += (size_t)gridDim.x * blockDim.x) {
    float4 acc = ((const float4*)parts)[i];
    for (int s = 1; s < ksplit; ++s) {
      float4 v = ((const float4*)(parts + (size_t)s * S))[i];
      acc.x += v.x; acc.y += v.y; acc.z += v.z; acc.w += v.w;
    }
    float4 o; o.x = -acc.x; o.y = -acc.y; o.z = -acc.z; o.w = -acc.w;
    ((float4*)costT)[i] = o;
  }
}

// ============================================================
// Hungarian (Jonker-Volgenant with potentials), exact match to the
// reference numpy code. cost is [Q,T] per batch -> transposed=True,
// so C = [n=T rows][m=Q cols] == costT[b][t][q] (contiguous rows).
// One block per batch; 512 threads parallelize minv-relax / argmin /
// potential updates. State lives in LDS.
// ============================================================
#define NTHR 512

__global__ __launch_bounds__(NTHR)
void hungarian_kernel(const float* __restrict__ costT,
                      int* __restrict__ outPred,
                      int* __restrict__ outTgt) {
  const int n = T_;        // rows (t), 200
  const int m = Q_;        // cols (q), 500

  __shared__ float v[Q_ + 1];
  __shared__ float minv[Q_ + 1];
  __shared__ float u[T_ + 1];
  __shared__ int   p[Q_ + 1];
  __shared__ int   way[Q_ + 1];
  __shared__ unsigned char used[Q_ + 1];
  __shared__ float rmin[NTHR];
  __shared__ int   ridx[NTHR];
  __shared__ int   sj0;

  const int t = threadIdx.x;
  const int b = blockIdx.x;
  const float* C = costT + (size_t)b * T_ * Q_;
  const float INF = __builtin_inff();

  if (t <= m) { v[t] = 0.f; p[t] = 0; way[t] = 0; }
  if (t <= n) u[t] = 0.f;
  __syncthreads();

  for (int i = 1; i <= n; ++i) {
    if (t == 0) { p[0] = i; sj0 = 0; }
    if (t <= m) { minv[t] = INF; used[t] = 0; }
    __syncthreads();

    while (true) {
      int j0 = sj0;
      int i0 = p[j0];
      if (t == j0) used[j0] = 1;
      __syncthreads();

      float ui0 = u[i0];
      if (t >= 1 && t <= m && !used[t]) {
        float cur = C[(size_t)(i0 - 1) * Q_ + (t - 1)] - ui0 - v[t];
        if (cur < minv[t]) { minv[t] = cur; way[t] = j0; }
        rmin[t] = minv[t];
      } else {
        rmin[t] = INF;
      }
      ridx[t] = t;
      __syncthreads();

      // arg-min with first-index tie-break (matches np.argmin)
      for (int s = NTHR / 2; s > 0; s >>= 1) {
        if (t < s) {
          float ov = rmin[t + s]; int oi = ridx[t + s];
          if (ov < rmin[t] || (ov == rmin[t] && oi < ridx[t])) { rmin[t] = ov; ridx[t] = oi; }
        }
        __syncthreads();
      }
      if (t == 0) sj0 = ridx[0];
      __syncthreads();

      float delta = rmin[0];
      int   j1    = sj0;
      if (t <= m) {
        if (used[t]) { u[p[t]] += delta; v[t] -= delta; }   // p[] distinct over used cols
        else if (t >= 1) minv[t] -= delta;
      }
      __syncthreads();
      if (p[j1] == 0) break;     // uniform: shared state, post-barrier
    }

    if (t == 0) {                // augment along 'way'
      int j0 = sj0;
      while (j0 != 0) { int j1 = way[j0]; p[j0] = p[j1]; j0 = j1; }
    }
    __syncthreads();
  }

  // Extract pairs in ascending q (== argsort(row_ind) of the reference).
  if (t == 0) {
    int cnt = 0;
    for (int j = 1; j <= m; ++j) {
      if (p[j] > 0) {
        outPred[(size_t)b * n + cnt] = j - 1;       // q index
        outTgt [(size_t)b * n + cnt] = p[j] - 1;    // t index
        ++cnt;
      }
    }
  }
}

// ============================================================
extern "C" void kernel_launch(void* const* d_in, const int* in_sizes, int n_in,
                              void* d_out, int out_size, void* d_ws, size_t ws_size,
                              hipStream_t stream) {
  const float* pred = (const float*)d_in[0];   // [B,Q,H,W] fp32
  const float* tgt  = (const float*)d_in[1];   // [B,T,H,W] fp32

  int* outPred = (int*)d_out;                  // [B,200] int32 (pred indices)
  int* outTgt  = outPred + B_ * T_;            // [B,200] int32 (tgt indices)

  // Workspace layout: [ksplit partial buffers][costT], each S bytes.
  // ksplit chosen from ws_size (constant per run -> deterministic).
  const size_t S = (size_t)B_ * T_ * Q_ * sizeof(float);   // 3.2 MB
  size_t navail = ws_size / S;
  int ksplit;
  float* parts = (float*)d_ws;
  float* costT;
  if (navail >= 2) {
    size_t k = navail - 1; if (k > 16) k = 16;
    ksplit = (int)k;
    costT  = (float*)((char*)d_ws + (size_t)ksplit * S);
  } else {
    ksplit = 1;
    costT  = (float*)d_ws;                     // in-place negate in reduction
  }

  // K-chunk per split, multiple of KT.
  int nkt    = HW_ / KT;                                   // 1024
  int kchunk = ((nkt + ksplit - 1) / ksplit) * KT;

  dim3 g((Q_ + MT - 1) / MT, (T_ + NT - 1) / NT, B_ * ksplit);  // up to (4,7,128)
  mask_cost_wmma<<<g, dim3(256), 0, stream>>>(pred, tgt, parts, ksplit, kchunk);

  reduce_negate<<<dim3(800), dim3(256), 0, stream>>>(parts, costT, ksplit);

  hungarian_kernel<<<dim3(B_), dim3(NTHR), 0, stream>>>(costT, outPred, outTgt);
}
